// NLI_MiniHeads_2173253451975
// MI455X (gfx1250) — matrix operand order repaired and checked by reading, not compiled, not run on hardware
//
#include <hip/hip_runtime.h>

// ---------------------------------------------------------------------------
// CDNA5 / gfx1250: bf16 WMMA implementation of the mini-head + trunk MLP.
//   1. transpose+convert all weights fp32 [K][N] -> bf16 [N][K] (Bt layout)
//   2. build "pairs" activations in bf16
//   3. fused mini-head kernel (64 rows/WG): GEMM1(1536->512)+lrelu+GEMM2
//      (512->64); the 64x512 intermediate lives in LDS; double-buffered
//      async global->LDS A tiles (ASYNCcnt pipeline)
//   4. generic WMMA GEMM for h1/h2/trunk with double-buffered async A+B
//      tiles; h2 writes straight into the [B,4096] LayerNorm layout
//   5. fp32 LayerNorm, then trunk GEMMs, then a tiny final dot (N=1)
// ---------------------------------------------------------------------------

typedef __attribute__((ext_vector_type(16))) __bf16 bf16x16;
typedef __attribute__((ext_vector_type(8)))  __bf16 bf16x8;
typedef __attribute__((ext_vector_type(8)))  float  v8f;

union FragU { bf16x16 v; bf16x8 h[2]; };

__device__ inline bf16x16 ld_frag(const __bf16* p0, const __bf16* p1) {
  FragU u;
  u.h[0] = *(const bf16x8*)p0;
  u.h[1] = *(const bf16x8*)p1;
  return u.v;
}

__device__ inline float lrelu(float v) { return v > 0.f ? v : 0.01f * v; }

// gfx1250 async global->LDS copy (16B per lane), tracked by ASYNCcnt.
// vdst = LDS byte address (low 32 bits of generic shared pointer), vaddr = VA.
__device__ inline void async_ld_b128(const void* gptr, void* lptr) {
  asm volatile("global_load_async_to_lds_b128 %0, %1, off"
               :: "v"((unsigned)(size_t)lptr), "v"(gptr)
               : "memory");
}
__device__ inline void wait_async() {
  asm volatile("s_wait_asynccnt 0x0" ::: "memory");
}

// ---------------------------------------------------------------------------
// Weight transpose + fp32->bf16 convert:  in [batch][K][N] -> out [batch][N][K]
// block (32,8), grid (N/32, K/32, batch). K,N are multiples of 32 here.
// ---------------------------------------------------------------------------
__global__ __launch_bounds__(256) void transpose_cvt(
    const float* __restrict__ in, __bf16* __restrict__ out, int K, int N) {
  __shared__ float tile[32][33];
  const int bt = blockIdx.z;
  const float* src = in  + (size_t)bt * K * N;
  __bf16*      dst = out + (size_t)bt * N * K;
  const int n0 = blockIdx.x * 32, k0 = blockIdx.y * 32;
  const int tx = threadIdx.x, ty = threadIdx.y;
#pragma unroll
  for (int i = 0; i < 4; ++i) {
    tile[ty + i * 8][tx] = src[(size_t)(k0 + ty + i * 8) * N + (n0 + tx)];
  }
  __syncthreads();
#pragma unroll
  for (int i = 0; i < 4; ++i) {
    dst[(size_t)(n0 + ty + i * 8) * K + (k0 + tx)] = (__bf16)tile[tx][ty + i * 8];
  }
}

// ---------------------------------------------------------------------------
// pairs[b*10+p][c] = c<768 ? emb[b][0][c] : emb[b][p+1][c-768]   (bf16)
// ---------------------------------------------------------------------------
__global__ __launch_bounds__(256) void build_pairs(
    const float* __restrict__ emb, __bf16* __restrict__ pairs) {
  const size_t idx = (size_t)blockIdx.x * 256 + threadIdx.x;  // < 20480*1536
  const int    c = (int)(idx % 1536);
  const size_t r = idx / 1536;
  const int    p = (int)(r % 10);
  const size_t b = r / 10;
  const float v = (c < 768) ? emb[(b * 11 + 0) * 768 + c]
                            : emb[(b * 11 + (p + 1)) * 768 + (c - 768)];
  pairs[idx] = (__bf16)v;
}

// ---------------------------------------------------------------------------
// Fused mini-head kernel: per (head, 64-row tile)
//   H  = lrelu(pairs[64x1536] @ W1[1536x512] + b1)   -> LDS (bf16)
//   C2 = H @ W2[512x64] + b2                          -> scatter to mini layout
// grid (20480/64, 32 heads), block 256 (8 wave32s).
// A tiles: double-buffered async global->LDS; W1/W2 frags direct from L2.
// ---------------------------------------------------------------------------
__global__ __launch_bounds__(256) void mini_fused(
    const __bf16* __restrict__ pairs, const __bf16* __restrict__ w1t,
    const float* __restrict__ b1, const __bf16* __restrict__ w2t,
    const float* __restrict__ b2, __bf16* __restrict__ mini) {
  __shared__ __bf16 sA[2][64 * 40]; // 64x32 A tiles, rows padded to 40 bf16
  __shared__ __bf16 sH[64 * 520];   // 64x512 intermediate, padded rows

  const int head = blockIdx.y;
  const int row0 = blockIdx.x * 64;
  const int tid  = threadIdx.x;
  const int w    = tid >> 5;        // wave id 0..7 (wave32)
  const int lane = tid & 31;
  const int hf   = lane >> 4;       // lane half
  const int li   = lane & 15;

  const __bf16* W1 = w1t + (size_t)head * 512 * 1536;  // [512][1536]
  const __bf16* W2 = w2t + (size_t)head * 64 * 512;    // [64][512]

  // ---- stage A: 64x512 output; wave w owns columns [w*64, w*64+64) ----
  v8f acc[4][4] = {};
  {
    const int row = tid >> 2, ch = tid & 3;   // 64 rows x 4 x 16B chunks
    const int loff = row * 40 + ch * 8;
    const __bf16* gsrc = &pairs[(size_t)(row0 + row) * 1536 + ch * 8];
    async_ld_b128(gsrc, &sA[0][loff]);        // prime the pipeline
    for (int it = 0; it < 48; ++it) {
      const int k0  = it * 32;
      const int cur = it & 1;
      wait_async();
      __syncthreads();
      if (it + 1 < 48) async_ld_b128(gsrc + k0 + 32, &sA[cur ^ 1][loff]);
      bf16x16 afr[4], bfr[4];
#pragma unroll
      for (int rt = 0; rt < 4; ++rt) {
        const int ar = rt * 16 + li;
        afr[rt] = ld_frag(&sA[cur][ar * 40 + 8 * hf],
                          &sA[cur][ar * 40 + 16 + 8 * hf]);
      }
#pragma unroll
      for (int ct = 0; ct < 4; ++ct) {
        const int n = w * 64 + ct * 16 + li;
        const __bf16* bp = &W1[(size_t)n * 1536 + k0 + 8 * hf];
        bfr[ct] = ld_frag(bp, bp + 16);
      }
#pragma unroll
      for (int rt = 0; rt < 4; ++rt)
#pragma unroll
        for (int ct = 0; ct < 4; ++ct)
          acc[rt][ct] = __builtin_amdgcn_wmma_f32_16x16x32_bf16(
              false, afr[rt], false, bfr[ct], (short)0, acc[rt][ct], false, false);
    }
  }
  __syncthreads();
  // epilogue A: bias + lrelu, write bf16 into sH
#pragma unroll
  for (int ct = 0; ct < 4; ++ct) {
    const int n = w * 64 + ct * 16 + li;
    const float bv = b1[head * 512 + n];
#pragma unroll
    for (int rt = 0; rt < 4; ++rt)
#pragma unroll
      for (int e = 0; e < 8; ++e) {
        const int m = rt * 16 + e + 8 * hf;
        sH[m * 520 + n] = (__bf16)lrelu(acc[rt][ct][e] + bv);
      }
  }
  __syncthreads();

  // ---- stage B: 64x64 output; two 16x16 tiles per wave ----
  const int rt2   = w >> 1;           // tiles 2w, 2w+1 share the row tile
  const int cbase = (w & 1) * 2;
  v8f acc2[2] = {};
  for (int k0 = 0; k0 < 512; k0 += 32) {
    const int ar = rt2 * 16 + li;
    bf16x16 a2 = ld_frag(&sH[ar * 520 + k0 + 8 * hf],
                         &sH[ar * 520 + k0 + 16 + 8 * hf]);
#pragma unroll
    for (int i = 0; i < 2; ++i) {
      const int n = (cbase + i) * 16 + li;
      const __bf16* bp = &W2[(size_t)n * 512 + k0 + 8 * hf];
      bf16x16 b2v = ld_frag(bp, bp + 16);
      acc2[i] = __builtin_amdgcn_wmma_f32_16x16x32_bf16(
          false, a2, false, b2v, (short)0, acc2[i], false, false);
    }
  }
  const int hh = head >> 2, m4 = head & 3;
#pragma unroll
  for (int i = 0; i < 2; ++i) {
    const int n = (cbase + i) * 16 + li;
    const float bv = b2[head * 64 + n];
#pragma unroll
    for (int e = 0; e < 8; ++e) {
      const int m = rt2 * 16 + e + 8 * hf;
      const int r = row0 + m;               // global pair-row in [0,20480)
      const int b = r / 10, p = r % 10;
      // mini[hh][b][m4*640 + p*64 + n]  (the reference's reshape, for free)
      mini[((size_t)hh * 2048 + b) * 2560 + m4 * 640 + p * 64 + n] =
          (__bf16)(acc2[i][e] + bv);
    }
  }
}

// ---------------------------------------------------------------------------
// Generic bf16 WMMA GEMM:  C = op(A @ Bt^T + bias)
//   A  row-major [M][K] bf16, Bt row-major [N][K] bf16, bias fp32 [N]
//   BM=64, BN=128, K-step 32; block 256 = 8 waves in a 2x4 grid of 32x32 tiles
//   Double-buffered async global->LDS copies for both A and B tiles.
//   OUT_MODE 0: bf16 row-major (ldc), batch offset z*sC
//   OUT_MODE 1: fp32 heads-scatter  C[m*4096 + z*512 + n]
// Requires M % 64 == 0, K % 32 == 0 (true for every stage here).
// ---------------------------------------------------------------------------
template <bool LRELU, int OUT_MODE>
__global__ __launch_bounds__(256) void gemm_nt(
    const __bf16* __restrict__ A, const __bf16* __restrict__ Bt,
    const float* __restrict__ bias, void* __restrict__ Cout,
    int M, int N, int K,
    long long sA, long long sB, long long sBias, long long sC, int ldc) {
  __shared__ __bf16 sAt[2][64 * 40];
  __shared__ __bf16 sBt[2][128 * 40];

  const int z = blockIdx.z;
  const __bf16* Ab = A    + (size_t)z * sA;
  const __bf16* Bb = Bt   + (size_t)z * sB;
  const float*  bb = bias + (size_t)z * sBias;

  const int by = blockIdx.x * 64;    // row block
  const int bx = blockIdx.y * 128;   // col block
  const int tid = threadIdx.x;
  const int w = tid >> 5, lane = tid & 31;
  const int hf = lane >> 4, li = lane & 15;
  const int wr = (w >> 2) * 32;      // wave row offset within block
  const int wc = (w & 3) * 32;       // wave col offset within block

  v8f acc[2][2] = {};
  const float4 zero4 = make_float4(0.f, 0.f, 0.f, 0.f);

  // per-thread tile-copy assignments
  const int arow = tid >> 2, ach = tid & 3;            // A: 64 rows x 4 chunks
  const int aoff = arow * 40 + ach * 8;
  const __bf16* agsrc = &Ab[(size_t)(by + arow) * K + ach * 8];
  const int brow0 = tid >> 1, bch0 = (tid & 1) * 2;    // B: 128 rows x 2x(2 ch)
  // Each thread copies B rows brow0 chunks {bch0, bch0+1}
  const int boff0 = brow0 * 40 + bch0 * 8;
  const int bn    = bx + brow0;
  const __bf16* bgsrc = &Bb[(size_t)bn * K + bch0 * 8];
  const bool bvalid = bn < N;

  // zero-fill out-of-range B rows once (both buffers); only hits in L4 stage
  if (!bvalid) {
#pragma unroll
    for (int d = 0; d < 2; ++d) {
      *(float4*)&sBt[d][boff0]     = zero4;
      *(float4*)&sBt[d][boff0 + 8] = zero4;
    }
  }

  const int nk = K >> 5;
  // prime the async pipeline for k0 = 0
  async_ld_b128(agsrc, &sAt[0][aoff]);
  if (bvalid) {
    async_ld_b128(bgsrc,      &sBt[0][boff0]);
    async_ld_b128(bgsrc + 8,  &sBt[0][boff0 + 8]);
  }

  for (int it = 0; it < nk; ++it) {
    const int cur = it & 1;
    wait_async();
    __syncthreads();
    if (it + 1 < nk) {
      const int kn = (it + 1) * 32;
      async_ld_b128(agsrc + kn, &sAt[cur ^ 1][aoff]);
      if (bvalid) {
        async_ld_b128(bgsrc + kn,      &sBt[cur ^ 1][boff0]);
        async_ld_b128(bgsrc + kn + 8,  &sBt[cur ^ 1][boff0 + 8]);
      }
    }
    bf16x16 afr[2], bfr[2];
#pragma unroll
    for (int rt = 0; rt < 2; ++rt) {
      const int ar = wr + rt * 16 + li;
      afr[rt] = ld_frag(&sAt[cur][ar * 40 + 8 * hf],
                        &sAt[cur][ar * 40 + 16 + 8 * hf]);
    }
#pragma unroll
    for (int ct = 0; ct < 2; ++ct) {
      const int br = wc + ct * 16 + li;
      bfr[ct] = ld_frag(&sBt[cur][br * 40 + 8 * hf],
                        &sBt[cur][br * 40 + 16 + 8 * hf]);
    }
#pragma unroll
    for (int rt = 0; rt < 2; ++rt)
#pragma unroll
      for (int ct = 0; ct < 2; ++ct)
        acc[rt][ct] = __builtin_amdgcn_wmma_f32_16x16x32_bf16(
            false, afr[rt], false, bfr[ct], (short)0, acc[rt][ct], false, false);
  }

#pragma unroll
  for (int ct = 0; ct < 2; ++ct) {
    const int n = bx + wc + ct * 16 + li;
    const float bv = (n < N) ? bb[n] : 0.f;
#pragma unroll
    for (int rt = 0; rt < 2; ++rt)
#pragma unroll
      for (int e = 0; e < 8; ++e) {
        const int m = by + wr + rt * 16 + e + 8 * hf;
        float v = acc[rt][ct][e] + bv;
        if (LRELU) v = lrelu(v);
        if (OUT_MODE == 0) {
          if (n < N)
            ((__bf16*)Cout + (size_t)z * sC)[(size_t)m * ldc + n] = (__bf16)v;
        } else {
          ((float*)Cout)[(size_t)m * 4096 + z * 512 + n] = v;
        }
      }
  }
}

// ---------------------------------------------------------------------------
// LayerNorm over 4096, fp32 in -> bf16 out. One row per block (256 threads).
// ---------------------------------------------------------------------------
__global__ __launch_bounds__(256) void layernorm_k(
    const float* __restrict__ x, const float* __restrict__ g,
    const float* __restrict__ beta, __bf16* __restrict__ out) {
  const int row = blockIdx.x, tid = threadIdx.x;
  const float* xr = x + (size_t)row * 4096;
  float vals[16], s = 0.f, sq = 0.f;
#pragma unroll
  for (int i = 0; i < 16; ++i) {
    const float v = xr[tid + i * 256];
    vals[i] = v; s += v; sq += v * v;
  }
  __shared__ float rs[256], rq[256];
  rs[tid] = s; rq[tid] = sq;
  __syncthreads();
  for (int off = 128; off > 0; off >>= 1) {
    if (tid < off) { rs[tid] += rs[tid + off]; rq[tid] += rq[tid + off]; }
    __syncthreads();
  }
  const float mu  = rs[0] * (1.f / 4096.f);
  const float var = rq[0] * (1.f / 4096.f) - mu * mu;
  const float inv = rsqrtf(var + 1e-5f);
#pragma unroll
  for (int i = 0; i < 16; ++i) {
    const int c = tid + i * 256;
    out[(size_t)row * 4096 + c] = (__bf16)((vals[i] - mu) * inv * g[c] + beta[c]);
  }
}

// ---------------------------------------------------------------------------
// Final layer: out[b] = t4[b,:64] . l5_w + l5_b   (fp32 out)
// ---------------------------------------------------------------------------
__global__ __launch_bounds__(256) void final_dot(
    const __bf16* __restrict__ t4, const float* __restrict__ w,
    const float* __restrict__ b, float* __restrict__ out) {
  const int r = blockIdx.x * 256 + threadIdx.x;
  if (r >= 2048) return;
  float acc = b[0];
#pragma unroll
  for (int k = 0; k < 64; ++k) acc += (float)t4[r * 64 + k] * w[k];
  out[r] = acc;
}

// ---------------------------------------------------------------------------
extern "C" void kernel_launch(void* const* d_in, const int* in_sizes, int n_in,
                              void* d_out, int out_size, void* d_ws, size_t ws_size,
                              hipStream_t stream) {
  (void)in_sizes; (void)n_in; (void)out_size; (void)ws_size;
  const float* emb  = (const float*)d_in[0];
  const float* c1_w = (const float*)d_in[1];
  const float* c1_b = (const float*)d_in[2];
  const float* c2_w = (const float*)d_in[3];
  const float* c2_b = (const float*)d_in[4];
  const float* h1_w = (const float*)d_in[5];
  const float* h1_b = (const float*)d_in[6];
  const float* h2_w = (const float*)d_in[7];
  const float* h2_b = (const float*)d_in[8];
  const float* ln_g = (const float*)d_in[9];
  const float* ln_b = (const float*)d_in[10];
  const float* l1_w = (const float*)d_in[11];
  const float* l1_b = (const float*)d_in[12];
  const float* l2_w = (const float*)d_in[13];
  const float* l2_b = (const float*)d_in[14];
  const float* l3_w = (const float*)d_in[15];
  const float* l3_b = (const float*)d_in[16];
  const float* l4_w = (const float*)d_in[17];
  const float* l4_b = (const float*)d_in[18];
  const float* l5_w = (const float*)d_in[19];
  const float* l5_b = (const float*)d_in[20];
  float* out = (float*)d_out;

  // workspace carve-up (256B aligned); total ~346 MB
  char* p = (char*)d_ws;
  auto take = [&](size_t bytes) -> char* {
    char* r = p;
    p += (bytes + 255) & ~(size_t)255;
    return r;
  };
  __bf16* pairs  = (__bf16*)take((size_t)20480 * 1536 * 2);
  __bf16* c1_wt  = (__bf16*)take((size_t)32 * 512 * 1536 * 2);
  __bf16* c2_wt  = (__bf16*)take((size_t)32 * 64 * 512 * 2);
  __bf16* h1_wt  = (__bf16*)take((size_t)8 * 768 * 2560 * 2);
  __bf16* h2_wt  = (__bf16*)take((size_t)8 * 512 * 768 * 2);
  __bf16* l1_wt  = (__bf16*)take((size_t)2048 * 4096 * 2);
  __bf16* l2_wt  = (__bf16*)take((size_t)768 * 2048 * 2);
  __bf16* l3_wt  = (__bf16*)take((size_t)256 * 768 * 2);
  __bf16* l4_wt  = (__bf16*)take((size_t)64 * 256 * 2);
  __bf16* mini   = (__bf16*)take((size_t)8 * 2048 * 2560 * 2);
  __bf16* xh     = (__bf16*)take((size_t)8 * 2048 * 768 * 2);
  float*  xf32   = (float*) take((size_t)2048 * 4096 * 4);
  __bf16* xln    = (__bf16*)take((size_t)2048 * 4096 * 2);
  __bf16* t1     = (__bf16*)take((size_t)2048 * 2048 * 2);
  __bf16* t2     = (__bf16*)take((size_t)2048 * 768 * 2);
  __bf16* t3     = (__bf16*)take((size_t)2048 * 256 * 2);
  __bf16* t4     = (__bf16*)take((size_t)2048 * 64 * 2);

  const dim3 tb(32, 8);
  // weight transpose+convert: in [batch][K][N] -> out [batch][N][K]
  transpose_cvt<<<dim3(512 / 32, 1536 / 32, 32), tb, 0, stream>>>(c1_w, c1_wt, 1536, 512);
  transpose_cvt<<<dim3(64 / 32, 512 / 32, 32), tb, 0, stream>>>(c2_w, c2_wt, 512, 64);
  transpose_cvt<<<dim3(768 / 32, 2560 / 32, 8), tb, 0, stream>>>(h1_w, h1_wt, 2560, 768);
  transpose_cvt<<<dim3(512 / 32, 768 / 32, 8), tb, 0, stream>>>(h2_w, h2_wt, 768, 512);
  transpose_cvt<<<dim3(2048 / 32, 4096 / 32, 1), tb, 0, stream>>>(l1_w, l1_wt, 4096, 2048);
  transpose_cvt<<<dim3(768 / 32, 2048 / 32, 1), tb, 0, stream>>>(l2_w, l2_wt, 2048, 768);
  transpose_cvt<<<dim3(256 / 32, 768 / 32, 1), tb, 0, stream>>>(l3_w, l3_wt, 768, 256);
  transpose_cvt<<<dim3(64 / 32, 256 / 32, 1), tb, 0, stream>>>(l4_w, l4_wt, 256, 64);

  build_pairs<<<(20480 * 1536) / 256, 256, 0, stream>>>(emb, pairs);

  // fused mini-head stage: 1.03 TFLOP of bf16 WMMA, no HBM intermediate
  mini_fused<<<dim3(20480 / 64, 32), 256, 0, stream>>>(pairs, c1_wt, c1_b, c2_wt, c2_b, mini);

  // per-head MLP
  gemm_nt<true, 0><<<dim3(32, 6, 8), 256, 0, stream>>>(
      mini, h1_wt, h1_b, xh, 2048, 768, 2560,
      (long long)2048 * 2560, (long long)768 * 2560, 768, (long long)2048 * 768, 768);
  gemm_nt<false, 1><<<dim3(32, 4, 8), 256, 0, stream>>>(
      xh, h2_wt, h2_b, xf32, 2048, 512, 768,
      (long long)2048 * 768, (long long)512 * 768, 512, 0, 4096);

  layernorm_k<<<2048, 256, 0, stream>>>(xf32, ln_g, ln_b, xln);

  // trunk
  gemm_nt<true, 0><<<dim3(32, 16, 1), 256, 0, stream>>>(
      xln, l1_wt, l1_b, t1, 2048, 2048, 4096, 0, 0, 0, 0, 2048);
  gemm_nt<true, 0><<<dim3(32, 6, 1), 256, 0, stream>>>(
      t1, l2_wt, l2_b, t2, 2048, 768, 2048, 0, 0, 0, 0, 768);
  gemm_nt<true, 0><<<dim3(32, 2, 1), 256, 0, stream>>>(
      t2, l3_wt, l3_b, t3, 2048, 256, 768, 0, 0, 0, 0, 256);
  gemm_nt<true, 0><<<dim3(32, 1, 1), 256, 0, stream>>>(
      t3, l4_wt, l4_b, t4, 2048, 64, 256, 0, 0, 0, 0, 64);

  final_dot<<<8, 256, 0, stream>>>(t4, l5_w, l5_b, out);
}
